// treelstm_27986006900813
// MI455X (gfx1250) — matrix operand order repaired and checked by reading, not compiled, not run on hardware
//
#include <hip/hip_runtime.h>
#include <hip/hip_bf16.h>
#include <stdint.h>

// ---------------------------------------------------------------------------
// TreeLSTM child-sum cell for MI455X (gfx1250, wave32, WMMA + TDM).
// Heavy op: sum_n sigmoid(child_h @ W_fh^T + b_fh) * child_c as a fused
// bf16-WMMA GEMM+reduce. B tiles are streamed into LDS by the Tensor Data
// Mover (tensor_load_to_lds, TENSORcnt) double-buffered ahead of compute, so
// the inner loop is ds_load_b128 (A) + strided 16-bit B loads + v_wmma_f32_16x16x32_bf16.
// ---------------------------------------------------------------------------

typedef __attribute__((ext_vector_type(16))) __bf16 v16bf;
typedef __attribute__((ext_vector_type(8)))  __bf16 v8bf;
typedef __attribute__((ext_vector_type(8)))  float  v8f;
typedef __attribute__((ext_vector_type(4)))  unsigned int v4u;
typedef __attribute__((ext_vector_type(8)))  int v8i;
typedef __attribute__((ext_vector_type(4)))  int v4i;

#define MEM_DIM      1024
#define N_CHILD      16384
#define ROWS_PER_WG  64                        // 4 row-tiles of 16
#define A_BYTES      (ROWS_PER_WG * MEM_DIM * 2)   // 128 KB bf16, fragment order
#define B_TILE_COLS  32                        // column-pair tile
#define B_TILE_ELEMS (B_TILE_COLS * MEM_DIM)   // 32768 bf16
#define B_TILE_BYTES (B_TILE_ELEMS * 2)        // 64 KB
#define N_JP         (MEM_DIM / B_TILE_COLS)   // 32 column-pair tiles
#define FC_OFF       (A_BYTES + 2 * B_TILE_BYTES)
#define SMEM_BYTES   (FC_OFF + MEM_DIM * 4)    // 260 KB of 320 KB WGP LDS

#if __has_builtin(__builtin_amdgcn_tensor_load_to_lds) && defined(__gfx1250__)
#define HAVE_TDM 1
#else
#define HAVE_TDM 0
#endif

__device__ __forceinline__ unsigned short f2bf(float f) {
  unsigned u = __float_as_uint(f);                 // fp32 -> bf16 RNE
  unsigned r = u + 0x7FFFu + ((u >> 16) & 1u);
  return (unsigned short)(r >> 16);
}
__device__ __forceinline__ float sigmoidf_(float x) {
  return 1.0f / (1.0f + __expf(-x));
}

// --------------------------- kernel 1: zero accumulator ---------------------
__global__ void zero_ws_kernel(float* __restrict__ p, int n) {
  int i = blockIdx.x * blockDim.x + threadIdx.x;
  if (i < n) p[i] = 0.0f;
}

// ----- kernel 2: W_fh [j][k] fp32 -> bf16 tile-contiguous Wt2 [jp][k][32] ---
__global__ void wfh_transpose_bf16_kernel(const float* __restrict__ W,
                                          unsigned short* __restrict__ Wt2) {
  __shared__ float tile[16][17];
  int tx = threadIdx.x, ty = threadIdx.y;
  int jb = blockIdx.x << 4, kb = blockIdx.y << 4;
  tile[ty][tx] = W[(size_t)(jb + ty) * MEM_DIM + (kb + tx)];   // coalesced read
  __syncthreads();
  int jo = jb + tx, ko = kb + ty;
  // element (k,j) lives at (j/32)*32768 + k*32 + (j%32): 64 KB per col-pair tile
  Wt2[((size_t)(jo >> 5) << 15) + (ko << 5) + (jo & 31)] = f2bf(tile[tx][ty]);
}

// --- TDM: async-copy one contiguous 64 KB B tile (Wt2 block jp) into LDS ----
__device__ __forceinline__ void tdm_load_b_tile(const unsigned short* Wt2,
                                                int jp, unsigned lds_off) {
#if HAVE_TDM
  unsigned long long ga =
      (unsigned long long)(uintptr_t)Wt2 + (unsigned long long)jp * B_TILE_BYTES;
  // D# group 0: count=1 valid, lds_addr, 57-bit global addr, type=2 ("image")
  v4u g0;
  g0[0] = 1u;
  g0[1] = lds_off;
  g0[2] = (unsigned)(ga & 0xffffffffu);
  g0[3] = (unsigned)(ga >> 32) | 0x80000000u;
  // D# group 1: 1-D tensor, data_size=8B, dim0 = tile_dim0 = 8192 units (64 KB)
  v8i g1;
  g1[0] = 3 << 16;                       // wg_mask=0 | data_size=3 (8B)
  g1[1] = (int)((B_TILE_BYTES / 8) << 16);  // tensor_dim0 lo16 (8192)
  g1[2] = 1 << 16;                       // tensor_dim0 hi16=0 | tensor_dim1=1
  g1[3] = (int)((B_TILE_BYTES / 8) << 16);  // tensor_dim1 hi16=0 | tile_dim0=8192
  g1[4] = 1;                             // tile_dim1=1 | tile_dim2=0
  g1[5] = B_TILE_BYTES / 8;              // tensor_dim0_stride lo32
  g1[6] = 0;
  g1[7] = 0;
  v4i g2 = {0, 0, 0, 0};
  v4i g3 = {0, 0, 0, 0};
  v8i gx = {0, 0, 0, 0, 0, 0, 0, 0};     // extra operand of the 6-arg builtin
  __builtin_amdgcn_tensor_load_to_lds(g0, g1, g2, g3, gx, 0);
#endif
}
__device__ __forceinline__ void tdm_wait() {
#if HAVE_TDM
  __builtin_amdgcn_s_wait_tensorcnt(0);
#endif
}

// ---------- kernel 3: fused f-gate GEMM + sigmoid + *child_c + reduce -------
// A fragment LDS order (ISA 16-bit A 16x32): lanes 0-15 (row m) hold
// K {k0..k0+7, k0+16..k0+23}; lanes 16-31 hold K {k0+8..k0+15, k0+24..k0+31}.
// B fragment (bf16 32x16): lane L <-> column N=L%16, 16 K values in A's order ->
// 16 strided bf16 LDS reads (64 B row stride) from the TDM-staged [k][32col] tile.
__global__ __launch_bounds__(256)
void fgate_gemm_reduce_kernel(const float* __restrict__ child_h,
                              const float* __restrict__ child_c,
                              const unsigned short* __restrict__ Wt2, // bf16 tiles
                              const float* __restrict__ b_fh,
                              float* __restrict__ fc_sum) {
  extern __shared__ char smem_raw[];
  unsigned short* As = (unsigned short*)smem_raw;
  unsigned short* Bs = (unsigned short*)(smem_raw + A_BYTES);
  float*          fcs = (float*)(smem_raw + FC_OFF);
  const unsigned lds_base = (unsigned)(uintptr_t)smem_raw;   // LDS byte offset

  const int tid  = threadIdx.x;
  const int row0 = blockIdx.x * ROWS_PER_WG;

  // zero LDS fc partials
  for (int j = tid; j < MEM_DIM; j += 256) fcs[j] = 0.0f;

  // stage + convert 64 child_h rows into WMMA-A fragment order
  for (int i = tid; i < (ROWS_PER_WG * MEM_DIM / 4); i += 256) {
    int r = i >> 8;
    int k = (i & 255) << 2;
    float4 v = *(const float4*)(child_h + (size_t)(row0 + r) * MEM_DIM + k);
    int rt = r >> 4, m = r & 15;
    int ks = k >> 5, kk = k & 31;
    int g  = kk >> 3;
    int ln = m + ((g & 1) << 4);
    int e  = (kk & 7) + ((g >> 1) << 3);
    int off = (((rt << 5) + ks) << 9) + (ln << 4) + e;
    unsigned long long q =
        (unsigned long long)f2bf(v.x)
      | ((unsigned long long)f2bf(v.y) << 16)
      | ((unsigned long long)f2bf(v.z) << 32)
      | ((unsigned long long)f2bf(v.w) << 48);
    *(unsigned long long*)(&As[off]) = q;
  }

  const int wid  = tid >> 5;
  const int lane = tid & 31;
  const int m    = lane & 15;
  const int half = lane >> 4;
  const int rt   = wid & 3;        // row-tile owned by this wave
  const int ch   = wid >> 2;       // column half (16 cols) of the 32-col tile

  // prime the double buffer: B(0) completed, B(1) in flight
  if (wid == 0) {
    tdm_load_b_tile(Wt2, 0, lds_base + A_BYTES);
    tdm_wait();
    tdm_load_b_tile(Wt2, 1, lds_base + A_BYTES + B_TILE_BYTES);
  }
  __syncthreads();

  for (int jp = 0; jp < N_JP; ++jp) {
#if HAVE_TDM
    const unsigned short* B = Bs + (jp & 1) * B_TILE_ELEMS;   // LDS tile
#else
    const unsigned short* B = Wt2 + (size_t)jp * B_TILE_ELEMS; // global fallback
#endif
    v8f acc0 = {}, acc1 = {};
    const v8bf* Av = (const v8bf*)As;
    #pragma unroll 4
    for (int ks = 0; ks < 32; ks += 2) {
      // two independent accumulator chains to avoid WMMA->WMMA stalls
      union { v16bf v; v8bf h[2]; } a0, a1, b0, b1;
      int ab0 = (((rt << 5) + ks) << 6) + (lane << 1);
      a0.h[0] = Av[ab0];       a0.h[1] = Av[ab0 + 1];
      a1.h[0] = Av[ab0 + 64];  a1.h[1] = Av[ab0 + 65];
      const __bf16* Bb = (const __bf16*)B + (ks << 10) + (ch << 4) + m;
      #pragma unroll
      for (int s = 0; s < 16; ++s) {
        int kk = (s & 7) + (half << 3) + ((s >> 3) << 4);
        b0.v[s] = Bb[kk << 5];
        b1.v[s] = Bb[(kk + 32) << 5];
      }
      acc0 = __builtin_amdgcn_wmma_f32_16x16x32_bf16(
          false, a0.v, false, b0.v, (short)0, acc0, false, false);
      acc1 = __builtin_amdgcn_wmma_f32_16x16x32_bf16(
          false, a1.v, false, b1.v, (short)0, acc1, false, false);
    }
    // epilogue: bias + sigmoid, multiply by child_c, accumulate into LDS fc
    const int j = (jp << 5) + (ch << 4) + m;
    const float bias = b_fh[j];
    float psum = 0.0f;
    #pragma unroll
    for (int v = 0; v < 8; ++v) {
      int rloc = (rt << 4) + v + (half << 3);
      float f = sigmoidf_(acc0[v] + acc1[v] + bias);
      psum = fmaf(f, child_c[(size_t)(row0 + rloc) * MEM_DIM + j], psum);
    }
    psum += __shfl_xor(psum, 16, 32);     // lanes L and L+16 share a column
    if (lane < 16) atomicAdd(&fcs[j], psum);      // ds_add_f32

    __syncthreads();                      // all waves done with buf[jp&1]
    if (wid == 0) {
      tdm_wait();                         // B(jp+1) has landed
      if (jp + 2 < N_JP)
        tdm_load_b_tile(Wt2, jp + 2, lds_base + A_BYTES + (jp & 1) * B_TILE_BYTES);
    }
    __syncthreads();                      // B(jp+1) visible to all waves
  }

  // one global flush per WG (256-deep atomics per column)
  for (int j = tid; j < MEM_DIM; j += 256) atomicAdd(&fc_sum[j], fcs[j]);
}

// -------- kernel 4: i/u/o GEMVs (wave-per-output) + final c,h combine -------
__global__ __launch_bounds__(256)
void gates_combine_kernel(const float* __restrict__ hsum,
                          const float* __restrict__ Wih, const float* __restrict__ bih,
                          const float* __restrict__ Wuh, const float* __restrict__ buh,
                          const float* __restrict__ Woh, const float* __restrict__ boh,
                          const float* __restrict__ fc_sum,
                          float* __restrict__ out) {
  int gtid = blockIdx.x * blockDim.x + threadIdx.x;
  int j    = gtid >> 5;
  int lane = gtid & 31;
  if (j >= MEM_DIM) return;
  float si = 0.0f, su = 0.0f, so = 0.0f;
  for (int k = lane; k < MEM_DIM; k += 32) {
    float hv = hsum[k];
    si = fmaf(Wih[(size_t)j * MEM_DIM + k], hv, si);
    su = fmaf(Wuh[(size_t)j * MEM_DIM + k], hv, su);
    so = fmaf(Woh[(size_t)j * MEM_DIM + k], hv, so);
  }
  #pragma unroll
  for (int off = 16; off > 0; off >>= 1) {
    si += __shfl_down(si, off, 32);
    su += __shfl_down(su, off, 32);
    so += __shfl_down(so, off, 32);
  }
  if (lane == 0) {
    float ig = sigmoidf_(si + bih[j]);
    float u  = tanhf(su + buh[j]);
    float og = sigmoidf_(so + boh[j]);
    float c  = ig * u + fc_sum[j];
    out[j]           = c;
    out[MEM_DIM + j] = og * tanhf(c);
  }
}

// ---------------------------------------------------------------------------
extern "C" void kernel_launch(void* const* d_in, const int* in_sizes, int n_in,
                              void* d_out, int out_size, void* d_ws, size_t ws_size,
                              hipStream_t stream) {
  const float* child_c = (const float*)d_in[0];
  const float* child_h = (const float*)d_in[1];
  const float* hsum    = (const float*)d_in[2];
  const float* W_ih    = (const float*)d_in[3];
  const float* b_ih    = (const float*)d_in[4];
  const float* W_fh    = (const float*)d_in[5];
  const float* b_fh    = (const float*)d_in[6];
  const float* W_uh    = (const float*)d_in[7];
  const float* b_uh    = (const float*)d_in[8];
  const float* W_oh    = (const float*)d_in[9];
  const float* b_oh    = (const float*)d_in[10];
  float* out = (float*)d_out;

  // workspace: [0,4KB) fc_sum accumulator, [4KB, 4KB+2MB) bf16 tiled Wt2
  float*          fc_sum = (float*)d_ws;
  unsigned short* Wt2    = (unsigned short*)((char*)d_ws + 4096);

  zero_ws_kernel<<<dim3(4), dim3(256), 0, stream>>>(fc_sum, MEM_DIM);
  wfh_transpose_bf16_kernel<<<dim3(64, 64), dim3(16, 16), 0, stream>>>(W_fh, Wt2);
  fgate_gemm_reduce_kernel<<<dim3(N_CHILD / ROWS_PER_WG), dim3(256),
                             SMEM_BYTES, stream>>>(child_h, child_c, Wt2, b_fh,
                                                   fc_sum);
  gates_combine_kernel<<<dim3(128), dim3(256), 0, stream>>>(
      hsum, W_ih, b_ih, W_uh, b_uh, W_oh, b_oh, fc_sum, out);
}
